// GPT2SelfAttention_76132590289000
// MI455X (gfx1250) — compile-verified, matrix operand order checked by reading, not run on hardware
//
#include <hip/hip_runtime.h>

typedef __attribute__((ext_vector_type(16))) _Float16 v16h;
typedef __attribute__((ext_vector_type(8)))  _Float16 v8h;
typedef __attribute__((ext_vector_type(8)))  float    v8f;

#define S_LEN 1024
#define D_DIM 256
#define NBATCH 64

// Load a 16x32 f16 fragment (A-matrix layout, also used for B via B^T) from a
// row-major f16 matrix with leading dimension `ld` (elements). `p` points at
// the tile origin (row 0, k 0).
// ISA layout (16-bit A 16x32): lanes 0-15 hold M=0..15 with K halves
// {0..7, 16..23}; lanes 16-31 hold M=0..15 with K halves {8..15, 24..31}.
__device__ __forceinline__ v16h load_frag(const _Float16* __restrict__ p, int ld) {
    int lane = threadIdx.x & 31;
    int row  = lane & 15;
    int koff = (lane < 16) ? 0 : 8;
    const _Float16* base = p + row * ld + koff;
    v8h lo = *(const v8h*)(base);        // K = koff .. koff+7
    v8h hi = *(const v8h*)(base + 16);   // K = koff+16 .. koff+23
    v16h r;
#pragma unroll
    for (int i = 0; i < 8; ++i) { r[i] = lo[i]; r[8 + i] = hi[i]; }
    return r;
}

__device__ __forceinline__ v8f wmma_f16(v16h a, v16h b, v8f c) {
    return __builtin_amdgcn_wmma_f32_16x16x32_f16(false, a, false, b,
                                                  (short)0, c, false, false);
}

// ---------------------------------------------------------------------------
// Prep kernels
// ---------------------------------------------------------------------------
__global__ void cvt_f32_to_f16(const float* __restrict__ in,
                               _Float16* __restrict__ out, int n) {
    int i = blockIdx.x * blockDim.x + threadIdx.x;
    if (i < n) out[i] = (_Float16)in[i];
}

// in: f32 [K][N] row-major -> out: f16 [N][K] row-major (i.e. transposed)
__global__ void transpose_cvt(const float* __restrict__ in,
                              _Float16* __restrict__ out, int K, int N) {
    int i = blockIdx.x * blockDim.x + threadIdx.x;
    if (i < K * N) {
        int k = i / N, n = i - k * N;
        out[n * K + k] = (_Float16)in[i];
    }
}

// ---------------------------------------------------------------------------
// QKV projection: qkv[m][f] = sum_d xh[m][d] * Wqkv[d][f] + b[f]
// Each wave computes a 16x64 strip (4 accumulators) so the A fragment is
// reused across 4 WMMAs per k-step. Scatter to Q, K ([b][s][d] f16) and
// Vt ([b][d][s] f16, transposed for the PV GEMM B-fragment).
// ---------------------------------------------------------------------------
__global__ void qkv_gemm(const _Float16* __restrict__ xh,   // [65536][256]
                         const _Float16* __restrict__ wt,   // [768][256] = Wqkv^T
                         const float* __restrict__ bias,    // [768]
                         _Float16* __restrict__ Qh,
                         _Float16* __restrict__ Kh,
                         _Float16* __restrict__ Vt) {
    int wave = threadIdx.x >> 5;
    int lane = threadIdx.x & 31;
    int unit = blockIdx.x * 8 + wave;        // 4096 * 12 = 49152 units
    int mTile = unit / 12, ng = unit - mTile * 12;
    int mBase = mTile * 16, nBase0 = ng * 64;

    const _Float16* ap = xh + (size_t)mBase * D_DIM;
    v8f acc[4] = {};
#pragma unroll
    for (int k = 0; k < D_DIM; k += 32) {
        v16h a = load_frag(ap + k, D_DIM);
#pragma unroll
        for (int j = 0; j < 4; ++j) {
            v16h b = load_frag(wt + (size_t)(nBase0 + 16 * j) * D_DIM + k, D_DIM);
            acc[j] = wmma_f16(a, b, acc[j]);
        }
    }
    int n  = lane & 15;
    int mo = (lane < 16) ? 0 : 8;
#pragma unroll
    for (int j = 0; j < 4; ++j) {
        int col = nBase0 + 16 * j + n;
        float bv = bias[col];
        if (col < 256) {
#pragma unroll
            for (int r = 0; r < 8; ++r)
                Qh[(size_t)(mBase + mo + r) * D_DIM + col] =
                    (_Float16)(acc[j][r] + bv);
        } else if (col < 512) {
            int c = col - 256;
#pragma unroll
            for (int r = 0; r < 8; ++r)
                Kh[(size_t)(mBase + mo + r) * D_DIM + c] =
                    (_Float16)(acc[j][r] + bv);
        } else {
            int d = col - 512;
#pragma unroll
            for (int r = 0; r < 8; ++r) {
                int m  = mBase + mo + r;
                int bb = m >> 10, s = m & 1023;
                Vt[((size_t)bb * D_DIM + d) * S_LEN + s] =
                    (_Float16)(acc[j][r] + bv);
            }
        }
    }
}

// ---------------------------------------------------------------------------
// Fused attention: one block = (batch b, 16-row query tile qt).
// Phase 1: scores (causal tiles only) via WMMA, Q fragments hoisted into
//          registers (loop-invariant) -> LDS f32 [16][1024]
// Phase 2: row softmax (wave-per-2-rows), write attn f32 to d_out and P f16
//          to LDS [16][1024]
// Phase 3: out_head = P @ V via WMMA (K-loop clipped to causal extent),
//          store f16 to oh
// Dynamic LDS: 64KB scores + 32KB P = 96KB (<< 320KB/WGP).
// ---------------------------------------------------------------------------
__global__ void attn_kernel(const _Float16* __restrict__ Qh,
                            const _Float16* __restrict__ Kh,
                            const _Float16* __restrict__ Vt,
                            float* __restrict__ attn_out,   // [64][1024][1024]
                            _Float16* __restrict__ oh) {    // [65536][256]
    extern __shared__ char smem[];
    float*    sc = (float*)smem;                   // [16][1024]
    _Float16* pP = (_Float16*)(smem + 65536);      // [16][1024]

    int b  = blockIdx.x >> 6;
    int qt = blockIdx.x & 63;
    int qBase = qt * 16;
    int wave = threadIdx.x >> 5;
    int lane = threadIdx.x & 31;

    const _Float16* Qb = Qh + ((size_t)b * S_LEN + qBase) * D_DIM;
    const _Float16* Kb = Kh + (size_t)b * S_LEN * D_DIM;

    // Hoist the 8 Q k-step fragments (loop-invariant across key tiles)
    v16h qf[8];
#pragma unroll
    for (int ks = 0; ks < 8; ++ks) qf[ks] = load_frag(Qb + ks * 32, D_DIM);

    // Phase 1: causal score tiles nt = wave, wave+8, ... <= qt
    for (int nt = wave; nt <= qt; nt += 8) {
        const _Float16* kp = Kb + (size_t)nt * 16 * D_DIM;
        v8f acc = {};
#pragma unroll
        for (int ks = 0; ks < 8; ++ks) {
            v16h bb = load_frag(kp + ks * 32, D_DIM);
            acc = wmma_f16(qf[ks], bb, acc);
        }
        int n  = lane & 15;
        int mo = (lane < 16) ? 0 : 8;
#pragma unroll
        for (int r = 0; r < 8; ++r)
            sc[(mo + r) * S_LEN + nt * 16 + n] = acc[r] * 0.0625f; // 1/sqrt(256)
    }
    __syncthreads();

    // Phase 2: softmax; wave w owns rows 2w, 2w+1
    size_t attnRowBase = ((size_t)b * S_LEN + qBase) * S_LEN;
    for (int rr = 0; rr < 2; ++rr) {
        int row  = wave * 2 + rr;
        int qrow = qBase + row;
        float vbuf[32];
        float mx = -3.0e38f;
#pragma unroll
        for (int i = 0; i < 32; ++i) {
            int col = i * 32 + lane;
            float v = (col <= qrow) ? sc[row * S_LEN + col] : -10000.0f;
            vbuf[i] = v;
            mx = fmaxf(mx, v);
        }
#pragma unroll
        for (int off = 16; off; off >>= 1) mx = fmaxf(mx, __shfl_xor(mx, off, 32));
        float sum = 0.0f;
#pragma unroll
        for (int i = 0; i < 32; ++i) {
            int col = i * 32 + lane;
            float e = (col <= qrow) ? __expf(vbuf[i] - mx) : 0.0f;
            vbuf[i] = e;
            sum += e;
        }
#pragma unroll
        for (int off = 16; off; off >>= 1) sum += __shfl_xor(sum, off, 32);
        float inv = 1.0f / sum;
#pragma unroll
        for (int i = 0; i < 32; ++i) {
            int col = i * 32 + lane;
            float p = vbuf[i] * inv;
            attn_out[attnRowBase + (size_t)row * S_LEN + col] = p;
            pP[row * S_LEN + col] = (_Float16)p;
        }
    }
    __syncthreads();

    // Phase 3: out_head[16][256] = P[16][<=1024] @ V ; wave w owns n-tiles 2w, 2w+1
    int ksteps = (qt >> 1) + 1;   // ceil(16*(qt+1)/32); trailing P cols are 0
    const _Float16* Vb = Vt + (size_t)b * D_DIM * S_LEN;
    for (int t = 0; t < 2; ++t) {
        int nBase = (wave * 2 + t) * 16;
        v8f acc = {};
        for (int ks = 0; ks < ksteps; ++ks) {
            v16h a  = load_frag(pP + ks * 32, S_LEN);
            v16h bb = load_frag(Vb + (size_t)nBase * S_LEN + ks * 32, S_LEN);
            acc = wmma_f16(a, bb, acc);
        }
        int n  = lane & 15;
        int mo = (lane < 16) ? 0 : 8;
#pragma unroll
        for (int r = 0; r < 8; ++r)
            oh[((size_t)b * S_LEN + qBase + mo + r) * D_DIM + nBase + n] =
                (_Float16)acc[r];
    }
}

// ---------------------------------------------------------------------------
// Output projection: out[m][n] = sum_d oh[m][d] * Wout[d][n] + b_out[n]
// Each wave computes a 16x64 strip (4 accumulators, A-frag reuse).
// ---------------------------------------------------------------------------
__global__ void out_gemm(const _Float16* __restrict__ oh,   // [65536][256]
                         const _Float16* __restrict__ wot,  // [256][256] = Wout^T
                         const float* __restrict__ bias,    // [256]
                         float* __restrict__ out) {         // [65536][256] f32
    int wave = threadIdx.x >> 5;
    int lane = threadIdx.x & 31;
    int unit = blockIdx.x * 8 + wave;        // 4096 * 4 = 16384 units
    int mTile = unit >> 2, ng = unit & 3;
    int mBase = mTile * 16, nBase0 = ng * 64;

    const _Float16* ap = oh + (size_t)mBase * D_DIM;
    v8f acc[4] = {};
#pragma unroll
    for (int k = 0; k < D_DIM; k += 32) {
        v16h a = load_frag(ap + k, D_DIM);
#pragma unroll
        for (int j = 0; j < 4; ++j) {
            v16h b = load_frag(wot + (size_t)(nBase0 + 16 * j) * D_DIM + k, D_DIM);
            acc[j] = wmma_f16(a, b, acc[j]);
        }
    }
    int n  = lane & 15;
    int mo = (lane < 16) ? 0 : 8;
#pragma unroll
    for (int j = 0; j < 4; ++j) {
        int col = nBase0 + 16 * j + n;
        float bv = bias[col];
#pragma unroll
        for (int r = 0; r < 8; ++r)
            out[(size_t)(mBase + mo + r) * D_DIM + col] = acc[j][r] + bv;
    }
}

// ---------------------------------------------------------------------------
extern "C" void kernel_launch(void* const* d_in, const int* in_sizes, int n_in,
                              void* d_out, int out_size, void* d_ws, size_t ws_size,
                              hipStream_t stream) {
    const float* x    = (const float*)d_in[0];   // [64][1024][256]
    const float* Wqkv = (const float*)d_in[1];   // [256][768]
    const float* bqkv = (const float*)d_in[2];   // [768]
    const float* Wout = (const float*)d_in[3];   // [256][256]
    const float* bout = (const float*)d_in[4];   // [256]

    float* out  = (float*)d_out;                         // 65536*256 floats
    float* attn = out + (size_t)NBATCH * S_LEN * D_DIM;  // 64*1024*1024 floats

    char* ws = (char*)d_ws;
    const size_t MB32 = (size_t)32 << 20;
    _Float16* Qh    = (_Float16*)(ws);
    _Float16* Kh    = (_Float16*)(ws + MB32);
    _Float16* Vt    = (_Float16*)(ws + 2 * MB32);
    _Float16* xh    = (_Float16*)(ws + 3 * MB32);  // reused as oh after QKV
    _Float16* Wqkvt = (_Float16*)(ws + 4 * MB32);
    _Float16* Woutt = (_Float16*)(ws + 4 * MB32 + ((size_t)512 << 10));

    const int nx = NBATCH * S_LEN * D_DIM;  // 16,777,216

    cvt_f32_to_f16<<<nx / 256, 256, 0, stream>>>(x, xh, nx);
    transpose_cvt<<<(256 * 768) / 256, 256, 0, stream>>>(Wqkv, Wqkvt, 256, 768);
    transpose_cvt<<<(256 * 256) / 256, 256, 0, stream>>>(Wout, Woutt, 256, 256);

    // 4096 m-tiles * 12 n-groups = 49152 wave-units, 8 waves/block
    qkv_gemm<<<49152 / 8, 256, 0, stream>>>(xh, Wqkvt, bqkv, Qh, Kh, Vt);

    // 64 batches * 64 query tiles; 96KB dynamic LDS
    attn_kernel<<<NBATCH * 64, 256, 98304, stream>>>(Qh, Kh, Vt, attn, xh);

    // 4096 m-tiles * 4 n-groups = 16384 wave-units
    out_gemm<<<16384 / 8, 256, 0, stream>>>(xh, Woutt, bout, out);
}
